// ToroidalAttention_23708219474636
// MI455X (gfx1250) — matrix operand order repaired and checked by reading, not compiled, not run on hardware
//
#include <hip/hip_runtime.h>
#include <hip/hip_bf16.h>

typedef __attribute__((ext_vector_type(16))) _Float16 v16h;
typedef __attribute__((ext_vector_type(8)))  _Float16 v8h;
typedef __attribute__((ext_vector_type(8)))  float    v8f;

#define DIMD   1024
#define NSEQ   1024
#define BATCH  4
#define HEADS  16
#define DHEAD  64
#define MROWS  (BATCH * NSEQ)   // 4096

static __device__ __forceinline__ v16h cat8(v8h lo, v8h hi) {
  return __builtin_shufflevector(lo, hi, 0,1,2,3,4,5,6,7,8,9,10,11,12,13,14,15);
}

// 16B async copy global -> LDS (GLOBAL_LOAD_ASYNC_TO_LDS_B128, ASYNCcnt)
static __device__ __forceinline__ void async_cp16(const _Float16* g, _Float16* l) {
  unsigned int loff = (unsigned int)(unsigned long long)l;  // LDS byte offset
  asm volatile("global_load_async_to_lds_b128 %0, %1, off" :: "v"(loff), "v"(g) : "memory");
}
static __device__ __forceinline__ void async_wait0() {
  asm volatile("s_wait_asynccnt 0" ::: "memory");
}

static __device__ __forceinline__ float lane16_max(float v) {
#pragma unroll
  for (int off = 1; off < 16; off <<= 1) v = fmaxf(v, __shfl_xor(v, off, 32));
  return v;
}
static __device__ __forceinline__ float lane16_sum(float v) {
#pragma unroll
  for (int off = 1; off < 16; off <<= 1) v += __shfl_xor(v, off, 32);
  return v;
}
static __device__ __forceinline__ float wave_sum(float v) {
#pragma unroll
  for (int off = 1; off < 32; off <<= 1) v += __shfl_xor(v, off, 32);
  return v;
}

// ---------------------------------------------------------------- elementwise
__global__ void k_f32_to_f16(const float* __restrict__ src,
                             _Float16* __restrict__ dst, int n) {
  int i = blockIdx.x * blockDim.x + threadIdx.x;
  if (i < n) dst[i] = (_Float16)src[i];
}

// toroidal row-norm: toroidal_norm(t) == normalize(scale .* t), scale=[2]*512++[1]*512
__global__ void k_rownorm(const float* __restrict__ W, _Float16* __restrict__ Wh) {
  const int row  = blockIdx.x * 8 + (threadIdx.x >> 5);
  const int lane = threadIdx.x & 31;
  const float* src = W + (size_t)row * DIMD;
  float vals[32];
  float ss = 0.f;
#pragma unroll
  for (int i = 0; i < 32; ++i) {
    int c = lane + i * 32;
    float t = src[c] * ((c < DIMD / 2) ? 2.0f : 1.0f);
    vals[i] = t;
    ss += t * t;
  }
  ss = wave_sum(ss);
  float inv = rsqrtf(fmaxf(ss, 1e-20f));
  _Float16* dst = Wh + (size_t)row * DIMD;
#pragma unroll
  for (int i = 0; i < 32; ++i) dst[lane + i * 32] = (_Float16)(vals[i] * inv);
}

// toroidal column-norm of w_o (axis 0), output f16 same (D x E) layout
__global__ void k_colnorm(const float* __restrict__ W, _Float16* __restrict__ Wh) {
  const int col  = blockIdx.x * 8 + (threadIdx.x >> 5);
  const int lane = threadIdx.x & 31;
  float vals[32];
  float ss = 0.f;
#pragma unroll
  for (int i = 0; i < 32; ++i) {
    int r = lane + i * 32;
    float t = W[(size_t)r * DIMD + col] * ((r < DIMD / 2) ? 2.0f : 1.0f);
    vals[i] = t;
    ss += t * t;
  }
  ss = wave_sum(ss);
  float inv = rsqrtf(fmaxf(ss, 1e-20f));
#pragma unroll
  for (int i = 0; i < 32; ++i)
    Wh[(size_t)(lane + i * 32) * DIMD + col] = (_Float16)(vals[i] * inv);
}

// per-head toroidal norm over 64 channels; optional fused qk scale * attn scale
__global__ void k_headnorm(_Float16* __restrict__ T,
                           const float* __restrict__ qk, int apply) {
  const int idx  = blockIdx.x * 8 + (threadIdx.x >> 5);
  const int lane = threadIdx.x & 31;
  const int m = idx >> 4;
  const int h = idx & 15;
  _Float16* p = T + (size_t)m * DIMD + h * DHEAD;
  float s0 = (float)p[lane] * 2.0f;
  float s1 = (float)p[lane + 32];
  float ss = wave_sum(s0 * s0 + s1 * s1);
  float inv = rsqrtf(fmaxf(ss, 1e-20f));
  float r0 = s0 * inv, r1 = s1 * inv;
  if (apply) {                            // s = param * DIM, fused with sqrt(DH)=8
    r0 *= qk[h * DHEAD + lane]      * 1024.0f * 8.0f;
    r1 *= qk[h * DHEAD + lane + 32] * 1024.0f * 8.0f;
  }
  p[lane]      = (_Float16)r0;
  p[lane + 32] = (_Float16)r1;
}

// ---------------------------------------------------------------- WMMA GEMM
// C[m,n] = sum_k A[m,k]*W[n,k].  A: 4096x1024 f16, W: 1024x1024 f16 (N x K).
// Block: 8 waves -> 128x64 tile, wave -> 32x32, K-step 32.
// A tile async-staged to LDS; W tile transposed to LDS.
template <int OUTF32>
__global__ __launch_bounds__(256) void k_gemm(const _Float16* __restrict__ A,
                                              const _Float16* __restrict__ W,
                                              _Float16* __restrict__ Ch,
                                              float* __restrict__ Cf) {
  __shared__ _Float16 At[128][32];       // [m][k] tile, 64B rows (async staged)
  __shared__ _Float16 Wt[32][72];        // [k][n] transposed tile, 144B rows
  const int tid  = threadIdx.x;
  const int lane = tid & 31;
  const int wave = tid >> 5;
  const int wm = wave & 3;
  const int wn = wave >> 2;
  const int bm  = blockIdx.x * 128;
  const int bn  = blockIdx.y * 64;
  const int m0  = bm + wm * 32;
  const int n0w = wn * 32;
  const int K = DIMD, N = DIMD;

  v8f acc[2][2] = {};
  const int koff  = (lane < 16) ? 0 : 8;   // A frag: K {0-7,16-23} vs {8-15,24-31}
  const int stg_n = tid >> 2;              // 0..63
  const int stg_q = tid & 3;               // 0..3 (8 halves each)

  for (int k0 = 0; k0 < K; k0 += 32) {
    __syncthreads();                        // prior-iteration reads done
    // async-stage A tile: 128x32 halves = 512 x 16B units, 2 per thread
#pragma unroll
    for (int i = 0; i < 2; ++i) {
      int u = tid + i * 256;
      int row = u >> 2, q = u & 3;
      async_cp16(A + (size_t)(bm + row) * K + k0 + q * 8, &At[row][q * 8]);
    }
    // stage W tile transposed (global -> regs -> ds_store)
    {
      const _Float16* src = W + (size_t)(bn + stg_n) * K + k0 + stg_q * 8;
      if (k0 + 32 < K) __builtin_prefetch(src + 32, 0, 1);   // global_prefetch_b8
      v8h wv = *(const v8h*)src;
#pragma unroll
      for (int j = 0; j < 8; ++j) Wt[stg_q * 8 + j][stg_n] = wv[j];
    }
    async_wait0();                          // own async copies landed in LDS
    __syncthreads();                        // all staging visible block-wide

    v16h af[2], bf[2];
#pragma unroll
    for (int ms = 0; ms < 2; ++ms) {
      const _Float16* ap = &At[wm * 32 + ms * 16 + (lane & 15)][koff];
      af[ms] = cat8(*(const v8h*)ap, *(const v8h*)(ap + 16));
    }
#pragma unroll
    for (int ns = 0; ns < 2; ++ns) {       // B frag: lane = n, halves = k (table order)
      const int nn = n0w + ns * 16 + (lane & 15);
#pragma unroll
      for (int j = 0; j < 8; ++j) {
        bf[ns][j]     = Wt[koff + j][nn];
        bf[ns][8 + j] = Wt[16 + koff + j][nn];
      }
    }
#pragma unroll
    for (int ms = 0; ms < 2; ++ms)
#pragma unroll
      for (int ns = 0; ns < 2; ++ns)
        acc[ms][ns] = __builtin_amdgcn_wmma_f32_16x16x32_f16(
            false, af[ms], false, bf[ns], (short)0, acc[ms][ns], false, false);
  }
  const int rowhi = (lane >> 4) << 3;
#pragma unroll
  for (int ms = 0; ms < 2; ++ms)
#pragma unroll
    for (int ns = 0; ns < 2; ++ns)
#pragma unroll
      for (int r = 0; r < 8; ++r) {
        int m = m0 + ms * 16 + r + rowhi;
        int n = bn + n0w + ns * 16 + (lane & 15);
        if (OUTF32) Cf[(size_t)m * N + n] = acc[ms][ns][r];
        else        Ch[(size_t)m * N + n] = (_Float16)acc[ms][ns][r];
      }
}

// ---------------------------------------------------------------- attention
// grid (NSEQ/128, B*H); 8 waves/block, each wave owns a 16-row Q tile.
__global__ __launch_bounds__(256) void k_attn(const _Float16* __restrict__ Q,
                                              const _Float16* __restrict__ Kmat,
                                              const _Float16* __restrict__ V,
                                              _Float16* __restrict__ O) {
  __shared__ _Float16 Kt[64][40];          // [d][key] transposed K tile, 80B rows
  __shared__ _Float16 Vt[32][64];          // [key][d] V tile, async staged
  __shared__ _Float16 Pb[8][16][40];       // per-wave P staging (C -> A layout)
  const int tid = threadIdx.x, lane = tid & 31, wave = tid >> 5;
  const int b = blockIdx.y >> 4, h = blockIdx.y & 15;
  const int qrow0 = blockIdx.x * 128 + wave * 16;
  const size_t rowbase = (size_t)b * NSEQ;
  const int ecol = h * DHEAD;
  const int koff = (lane < 16) ? 0 : 8;
  const int arow = qrow0 + (lane & 15);
  const int rowhi = (lane >> 4) << 3;

  v16h qf[2];
#pragma unroll
  for (int s = 0; s < 2; ++s) {
    const _Float16* qp = Q + (rowbase + arow) * DIMD + ecol + s * 32 + koff;
    qf[s] = cat8(*(const v8h*)qp, *(const v8h*)(qp + 16));
  }

  v8f o[4] = {};
  float mrow[8], lrow[8];
#pragma unroll
  for (int r = 0; r < 8; ++r) { mrow[r] = -1e30f; lrow[r] = 0.f; }

  const int kend  = blockIdx.x * 128 + 128;
  const int stg_m = tid >> 3;   // 0..31 keys
  const int stg_q = tid & 7;    // 8-half chunks of the 64-wide head

  for (int kb = 0; kb < kend; kb += 32) {
    __syncthreads();
    // async-stage shared V tile: 32x64 halves = 256 x 16B, one per thread
    async_cp16(V + (rowbase + kb + stg_m) * DIMD + ecol + stg_q * 8,
               &Vt[stg_m][stg_q * 8]);
    // stage K tile transposed
    {
      const _Float16* kp = Kmat + (rowbase + kb + stg_m) * DIMD + ecol + stg_q * 8;
      v8h kv = *(const v8h*)kp;
#pragma unroll
      for (int j = 0; j < 8; ++j) Kt[stg_q * 8 + j][stg_m] = kv[j];
    }
    async_wait0();
    __syncthreads();
    if (kb > qrow0 + 15) continue;   // wave-uniform causal skip (barriers done)

    // scores: two 16x16 key tiles, contraction over d (2 x k32)
    v8f s2[2] = {};
#pragma unroll
    for (int ns = 0; ns < 2; ++ns)
#pragma unroll
      for (int ks = 0; ks < 2; ++ks) {
        v16h bf;
#pragma unroll
        for (int j = 0; j < 8; ++j) {
          bf[j]     = Kt[ks * 32 + koff + j][ns * 16 + (lane & 15)];
          bf[8 + j] = Kt[ks * 32 + 16 + koff + j][ns * 16 + (lane & 15)];
        }
        s2[ns] = __builtin_amdgcn_wmma_f32_16x16x32_f16(
            false, qf[ks], false, bf, (short)0, s2[ns], false, false);
      }

    // online softmax (row stats via 16-lane shuffles), probs -> LDS
#pragma unroll
    for (int r = 0; r < 8; ++r) {
      int qr = qrow0 + rowhi + r;
      int key0 = kb + (lane & 15);
      float v0 = (key0      <= qr) ? s2[0][r] : -1e30f;
      float v1 = (key0 + 16 <= qr) ? s2[1][r] : -1e30f;
      float mx = lane16_max(fmaxf(v0, v1));
      float mnew = fmaxf(mrow[r], mx);
      float corr = __expf(mrow[r] - mnew);
      float p0 = __expf(v0 - mnew);
      float p1 = __expf(v1 - mnew);
      float psum = lane16_sum(p0 + p1);
      lrow[r] = lrow[r] * corr + psum;
      mrow[r] = mnew;
#pragma unroll
      for (int t = 0; t < 4; ++t) o[t][r] *= corr;
      _Float16* prow = &Pb[wave][rowhi + r][0];
      prow[(lane & 15)]      = (_Float16)p0;
      prow[16 + (lane & 15)] = (_Float16)p1;
    }
    __threadfence_block();   // order per-wave ds stores before ds loads

    // P as A-fragment (16 q-rows x 32 keys)
    const _Float16* pp = &Pb[wave][lane & 15][koff];
    v16h pf = cat8(*(const v8h*)pp, *(const v8h*)(pp + 16));

    // V B-fragments from the shared LDS tile: lane = d, halves = key (table order)
#pragma unroll
    for (int t = 0; t < 4; ++t) {
      v16h bf;
#pragma unroll
      for (int j = 0; j < 8; ++j) {
        bf[j]     = Vt[koff + j][t * 16 + (lane & 15)];
        bf[8 + j] = Vt[16 + koff + j][t * 16 + (lane & 15)];
      }
      o[t] = __builtin_amdgcn_wmma_f32_16x16x32_f16(
          false, pf, false, bf, (short)0, o[t], false, false);
    }
  }

#pragma unroll
  for (int t = 0; t < 4; ++t)
#pragma unroll
    for (int r = 0; r < 8; ++r) {
      int qr  = qrow0 + rowhi + r;
      int col = ecol + t * 16 + (lane & 15);
      O[(rowbase + qr) * DIMD + col] = (_Float16)(o[t][r] / lrow[r]);
    }
}

// ---------------------------------------------------------------- launch
extern "C" void kernel_launch(void* const* d_in, const int* in_sizes, int n_in,
                              void* d_out, int out_size, void* d_ws, size_t ws_size,
                              hipStream_t stream) {
  (void)in_sizes; (void)n_in; (void)out_size; (void)ws_size;
  const float* x  = (const float*)d_in[0];
  const float* wq = (const float*)d_in[1];
  const float* wk = (const float*)d_in[2];
  const float* wv = (const float*)d_in[3];
  const float* wo = (const float*)d_in[4];
  const float* qk = (const float*)d_in[5];
  float* out = (float*)d_out;

  char* ws = (char*)d_ws;                       // total use: 40 MB
  _Float16* xh  = (_Float16*)(ws);              // 8 MB (reused for attn output)
  _Float16* wqn = (_Float16*)(ws + ( 8u << 20));
  _Float16* wkn = (_Float16*)(ws + (10u << 20));
  _Float16* wvn = (_Float16*)(ws + (12u << 20));
  _Float16* won = (_Float16*)(ws + (14u << 20));
  _Float16* q   = (_Float16*)(ws + (16u << 20));
  _Float16* k   = (_Float16*)(ws + (24u << 20));
  _Float16* v   = (_Float16*)(ws + (32u << 20));
  _Float16* ao  = xh;

  k_f32_to_f16<<<(MROWS * DIMD) / 256, 256, 0, stream>>>(x, xh, MROWS * DIMD);
  k_rownorm<<<DIMD / 8, 256, 0, stream>>>(wq, wqn);
  k_rownorm<<<DIMD / 8, 256, 0, stream>>>(wk, wkn);
  k_rownorm<<<DIMD / 8, 256, 0, stream>>>(wv, wvn);
  k_colnorm<<<DIMD / 8, 256, 0, stream>>>(wo, won);

  dim3 gg(MROWS / 128, DIMD / 64);
  k_gemm<0><<<gg, 256, 0, stream>>>(xh, wqn, q, nullptr);
  k_gemm<0><<<gg, 256, 0, stream>>>(xh, wkn, k, nullptr);
  k_gemm<0><<<gg, 256, 0, stream>>>(xh, wvn, v, nullptr);

  k_headnorm<<<(MROWS * HEADS) / 8, 256, 0, stream>>>(q, qk, 1);
  k_headnorm<<<(MROWS * HEADS) / 8, 256, 0, stream>>>(k, qk, 0);

  k_attn<<<dim3(NSEQ / 128, BATCH * HEADS), 256, 0, stream>>>(q, k, v, ao);

  k_gemm<1><<<gg, 256, 0, stream>>>(ao, won, nullptr, out);
}
